// LSTMFromScratch_2302102470941
// MI455X (gfx1250) — compile-verified, matrix operand order checked by reading, not run on hardware
//
#include <hip/hip_runtime.h>
#include <hip/hip_bf16.h>

// Problem sizes (fixed by the reference)
#define T_SEQ  1024
#define BATCH  32
#define INPUT  512
#define HIDDEN 512
#define OUTPUT 512
#define GATES  (4 * HIDDEN)          // 2048
#define NWG    16                    // workgroups in the persistent scan
#define HPER   (HIDDEN / NWG)        // 32 hidden units per WG

typedef __attribute__((ext_vector_type(16))) __bf16    v16bf;
typedef __attribute__((ext_vector_type(8)))  float     v8f;
typedef __attribute__((ext_vector_type(4)))  unsigned  u32x4;
typedef __attribute__((ext_vector_type(8)))  int       i32x8;
typedef __attribute__((ext_vector_type(4)))  int       i32x4;

#if __has_builtin(__builtin_amdgcn_tensor_load_to_lds) && __has_builtin(__builtin_amdgcn_s_wait_tensorcnt)
#define HAVE_TDM 1
#else
#define HAVE_TDM 0
#endif

// ---- workspace layout (bytes). Requires ~206.2 MB of d_ws. ----
#define WS_XPROJ 0ull                                               // bf16 [T*B][GATES]
#define WS_XBF   (WS_XPROJ + (size_t)T_SEQ * BATCH * GATES * 2)     // bf16 [T*B][INPUT]
#define WS_HSB   (WS_XBF   + (size_t)T_SEQ * BATCH * INPUT * 2)     // bf16 [T*B][HIDDEN]
#define WS_WXBF  (WS_HSB   + (size_t)T_SEQ * BATCH * HIDDEN * 2)    // bf16 [GATES][INPUT]
#define WS_WHBF  (WS_WXBF  + (size_t)GATES * INPUT * 2)             // bf16 [GATES][HIDDEN]
#define WS_WHYBF (WS_WHBF  + (size_t)GATES * HIDDEN * 2)            // bf16 [OUTPUT][HIDDEN]
#define WS_HBUF0 (WS_WHYBF + (size_t)OUTPUT * HIDDEN * 2)           // bf16 [B][H]
#define WS_HBUF1 (WS_HBUF0 + (size_t)BATCH * HIDDEN * 2)
#define WS_CTRL  (WS_HBUF1 + (size_t)BATCH * HIDDEN * 2)            // u32 arrive

// Scan kernel dynamic-LDS layout (byte offsets from smem base = LDS offset 0)
#define SC_WHL_OFF  0u                          // 128*512 bf16 = 131072
#define SC_XPL_OFF  131072u                     // 4*32*32 bf16 = 8192
#define SC_GATE_OFF 139264u                     // 32*128 f32  = 16384
#define SC_SMEM_SZ  155648u

// ---------------------------------------------------------------------------
// Tensor Data Mover: 2D tile load Global -> LDS (D# per CDNA5 ISA)
// ---------------------------------------------------------------------------
#if HAVE_TDM
__device__ __forceinline__ void tdm_load_2d(unsigned lds_byte_off, const void* gaddr,
                                            unsigned tile_d0, unsigned tile_d1,
                                            unsigned tensor_d0, unsigned tensor_d1,
                                            unsigned d0_stride, unsigned data_size_log2) {
  unsigned long long ga = (unsigned long long)gaddr;
  u32x4 g0;
  g0[0] = 1u;                                           // count=1, user descriptor
  g0[1] = lds_byte_off;                                 // lds_addr
  g0[2] = (unsigned)(ga & 0xFFFFFFFFu);                 // global_addr[31:0]
  g0[3] = (unsigned)((ga >> 32) & 0x1FFFFFFu) | (2u << 30);  // global_addr[56:32] | type=2
  i32x8 g1;
  g1[0] = (int)(data_size_log2 << 16);                  // wg_mask=0, data_size
  g1[1] = (int)((tensor_d0 & 0xFFFFu) << 16);           // [63:48] = tensor_dim0[15:0]
  g1[2] = (int)(((tensor_d0 >> 16) & 0xFFFFu) | ((tensor_d1 & 0xFFFFu) << 16));
  g1[3] = (int)(((tensor_d1 >> 16) & 0xFFFFu) | ((tile_d0 & 0xFFFFu) << 16));
  g1[4] = (int)(tile_d1 & 0xFFFFu);                     // tile_dim1, tile_dim2=0
  g1[5] = (int)d0_stride;                               // tensor_dim0_stride[31:0]
  g1[6] = 0;                                            // stride[47:32], dim1_stride lo
  g1[7] = 0;
  i32x4 z4 = {0, 0, 0, 0};
#if __clang_major__ >= 23
  i32x8 z8 = {0, 0, 0, 0, 0, 0, 0, 0};
  __builtin_amdgcn_tensor_load_to_lds(g0, g1, z4, z4, z8, 0);
#else
  __builtin_amdgcn_tensor_load_to_lds(g0, g1, z4, z4, 0);
#endif
}
#endif

// ---------------------------------------------------------------------------
// WMMA fragment helpers (CDNA5 wave32 layouts, ISA 7.12.2)
// A (16x32 bf16): lane L<16 -> M=L, K {0..7,16..23}; lane L+16 -> M=L, K {8..15,24..31}
// B (32x16 bf16): mirror with lane = N (row-major weight rows are B columns)
// C/D (16x16 f32): VGPR r, lanes 0..15 -> M=r, N=lane; lanes 16..31 -> M=8+r
// ---------------------------------------------------------------------------
__device__ __forceinline__ v16bf load_frag_b(const __bf16* p, int ld) {
  const int lane = threadIdx.x & 31;
  const __bf16* row = p + (size_t)(lane & 15) * ld + (lane >> 4) * 8;
  v16bf out;
#pragma unroll
  for (int i = 0; i < 8; ++i) out[i]     = row[i];
#pragma unroll
  for (int i = 0; i < 8; ++i) out[8 + i] = row[16 + i];
  return out;
}

__device__ __forceinline__ v8f load_acc_bf16(const __bf16* p, int ld) {
  const int lane = threadIdx.x & 31;
  const int n = lane & 15, mh = lane >> 4;
  v8f c;
#pragma unroll
  for (int r = 0; r < 8; ++r) c[r] = (float)p[(size_t)(mh * 8 + r) * ld + n];
  return c;
}

__device__ __forceinline__ v8f wmma_bf16(v16bf a, v16bf b, v8f c) {
  return __builtin_amdgcn_wmma_f32_16x16x32_bf16(false, a, false, b, (short)0, c, false, false);
}

__device__ __forceinline__ float sigmoid_f(float x) { return 1.0f / (1.0f + __expf(-x)); }
__device__ __forceinline__ float tanh_f(float x) {
  float ax = fabsf(x);
  float e  = __expf(-2.0f * ax);
  return copysignf((1.0f - e) / (1.0f + e), x);
}

// ---------------------------------------------------------------------------
// prep: convert activations/weights to bf16, zero h buffers + barrier counter
// ---------------------------------------------------------------------------
__global__ void lstm_prep_kernel(const float* __restrict__ x, const float* __restrict__ Wx,
                                 const float* __restrict__ Wh, const float* __restrict__ Why,
                                 __bf16* __restrict__ xbf, __bf16* __restrict__ wxbf,
                                 __bf16* __restrict__ whbf, __bf16* __restrict__ whybf,
                                 __bf16* __restrict__ h0, __bf16* __restrict__ h1,
                                 unsigned* __restrict__ arrive) {
  const size_t i = blockIdx.x * (size_t)blockDim.x + threadIdx.x;
  const size_t stride = (size_t)gridDim.x * blockDim.x;
  for (size_t k = i; k < (size_t)T_SEQ * BATCH * INPUT; k += stride) xbf[k] = (__bf16)x[k];
  for (size_t k = i; k < (size_t)GATES * INPUT;  k += stride) wxbf[k]  = (__bf16)Wx[k];
  for (size_t k = i; k < (size_t)GATES * HIDDEN; k += stride) whbf[k]  = (__bf16)Wh[k];
  for (size_t k = i; k < (size_t)OUTPUT * HIDDEN; k += stride) whybf[k] = (__bf16)Why[k];
  for (size_t k = i; k < (size_t)BATCH * HIDDEN; k += stride) {
    h0[k] = (__bf16)0.0f; h1[k] = (__bf16)0.0f;
  }
  if (i == 0) *arrive = 0u;
}

// ---------------------------------------------------------------------------
// GEMM+bias, all-bf16 operands: out[M][N] = A[M][K] * W[N][K]^T + bias[N]
// 128 threads / 4 waves; wave w: rows [by*128 + 32w, +32) (2 M-tiles),
// cols [bx*64, +64) (4 N-tiles) -> 8 WMMAs per K-step per wave.
// B tile (64x32 bf16) staged in LDS, double-buffered via TDM.
// ---------------------------------------------------------------------------
__global__ __launch_bounds__(128) void gemm_bias_bf16(
    const __bf16* __restrict__ A, const __bf16* __restrict__ W,
    const float* __restrict__ bias, int K, int ldn,
    float* __restrict__ outf, __bf16* __restrict__ outb) {
  extern __shared__ char smem[];                // 2 * 64*32 bf16 = 8192 B at LDS offset 0
  __bf16* Bt = (__bf16*)smem;
  const int wave = threadIdx.x >> 5;
  const int lane = threadIdx.x & 31;
  const int m0 = blockIdx.y * 128 + wave * 32;  // 2 M-tiles: m0, m0+16
  const int n0 = blockIdx.x * 64;

  v8f acc[2][4];
#pragma unroll
  for (int j = 0; j < 4; ++j) {
    float bv = bias[n0 + j * 16 + (lane & 15)];
#pragma unroll
    for (int r = 0; r < 8; ++r) { acc[0][j][r] = bv; acc[1][j][r] = bv; }
  }

  const int nk = K / 32;
#if HAVE_TDM
  if (wave == 0)
    tdm_load_2d(0u, W + (size_t)n0 * K, 32u, 64u, 32u, 64u, (unsigned)K, 1u);
#endif
  for (int kb = 0; kb < nk; ++kb) {
    const int buf = kb & 1;
#if HAVE_TDM
    if (wave == 0) __builtin_amdgcn_s_wait_tensorcnt(0);
    __syncthreads();                                   // tile kb visible to all waves
    if (wave == 0 && kb + 1 < nk)
      tdm_load_2d((unsigned)(((kb + 1) & 1) * 4096),
                  W + (size_t)n0 * K + (size_t)(kb + 1) * 32,
                  32u, 64u, 32u, 64u, (unsigned)K, 1u);
#else
    __syncthreads();
    for (int idx = threadIdx.x; idx < 64 * 32; idx += 128) {
      int r = idx >> 5, c = idx & 31;
      Bt[buf * 2048 + idx] = W[(size_t)(n0 + r) * K + (size_t)kb * 32 + c];
    }
    __syncthreads();
#endif
    if (kb + 2 < nk)
      __builtin_prefetch(A + (size_t)m0 * K + (size_t)(kb + 2) * 32, 0, 3);

    // Batch all fragment loads, then issue the 8 WMMAs.
    const __bf16* Bb = Bt + buf * 2048;
    v16bf a0 = load_frag_b(A + (size_t)m0 * K + (size_t)kb * 32, K);
    v16bf a1 = load_frag_b(A + (size_t)(m0 + 16) * K + (size_t)kb * 32, K);
    v16bf b[4];
#pragma unroll
    for (int j = 0; j < 4; ++j) b[j] = load_frag_b(Bb + (size_t)(j * 16) * 32, 32);
#pragma unroll
    for (int j = 0; j < 4; ++j) {
      acc[0][j] = wmma_bf16(a0, b[j], acc[0][j]);
      acc[1][j] = wmma_bf16(a1, b[j], acc[1][j]);
    }
  }

  const int mh = lane >> 4, nn = lane & 15;
#pragma unroll
  for (int mi = 0; mi < 2; ++mi)
#pragma unroll
    for (int j = 0; j < 4; ++j)
#pragma unroll
      for (int r = 0; r < 8; ++r) {
        size_t m = (size_t)(m0 + mi * 16 + mh * 8 + r);
        size_t n = (size_t)(n0 + j * 16 + nn);
        if (outf) outf[m * ldn + n] = acc[mi][j][r];
        else      outb[m * ldn + n] = (__bf16)acc[mi][j][r];
      }
}

// ---------------------------------------------------------------------------
// Persistent LSTM scan. NWG=16 blocks x 128 threads. WG owns 32 hidden units.
// LDS: Wh slice 128x512 bf16 (TDM-loaded once) + xproj tile (TDM-prefetched
// one step ahead) + gate exchange buffer. Wave w computes gate block w.
// ---------------------------------------------------------------------------
__global__ __launch_bounds__(128) void lstm_scan_kernel(
    const __bf16* __restrict__ whbf,    // [2048][512] bf16
    const __bf16* __restrict__ xproj,   // [T*B][2048] bf16 (bias folded in)
    __bf16* __restrict__ hsb,           // [T][B][512] bf16
    __bf16* __restrict__ hbuf0, __bf16* __restrict__ hbuf1,
    unsigned* __restrict__ arrive,
    float* __restrict__ hout, float* __restrict__ cout) {
  extern __shared__ char smem[];
  __bf16* WhL    = (__bf16*)(smem + SC_WHL_OFF);   // [128][512]
  __bf16* xpL    = (__bf16*)(smem + SC_XPL_OFF);   // [4][32][32]
  float*  gatesL = (float*)(smem + SC_GATE_OFF);   // [32][128]

  const int tid  = threadIdx.x;
  const int wave = tid >> 5;                 // 0..3 == gate block f,i,g,o
  const int lane = tid & 31;
  const int hid0 = blockIdx.x * HPER;

  // ---- preload Wh slice (4 strided 32x512 tiles) + xproj tile for t=0 ----
#if HAVE_TDM
  if (wave == 0) {
#pragma unroll
    for (int g = 0; g < 4; ++g)
      tdm_load_2d(SC_WHL_OFF + (unsigned)g * 32768u,
                  whbf + (size_t)(g * HIDDEN + hid0) * HIDDEN,
                  512u, 32u, 512u, 32u, (unsigned)HIDDEN, 1u);
#pragma unroll
    for (int g = 0; g < 4; ++g)
      tdm_load_2d(SC_XPL_OFF + (unsigned)g * 2048u,
                  xproj + (size_t)(g * HIDDEN + hid0),
                  32u, 32u, 32u, 32u, (unsigned)GATES, 1u);
    __builtin_amdgcn_s_wait_tensorcnt(0);
  }
  __syncthreads();
#else
  for (int idx = tid; idx < 128 * HIDDEN; idx += 128) {
    int row = idx >> 9, col = idx & (HIDDEN - 1);
    WhL[idx] = whbf[(size_t)(((row >> 5) << 9) + hid0 + (row & 31)) * HIDDEN + col];
  }
  for (int idx = tid; idx < 4 * 32 * 32; idx += 128) {
    int g = idx >> 10, b = (idx >> 5) & 31, u = idx & 31;
    xpL[idx] = xproj[(size_t)b * GATES + g * HIDDEN + hid0 + u];
  }
  __syncthreads();
#endif

  float c[8];
#pragma unroll
  for (int i = 0; i < 8; ++i) c[i] = 0.0f;
  const int uj    = tid & 31;                // owned hidden unit (local)
  const int bquad = tid >> 5;                // batches bquad*8 .. +7

  for (int t = 0; t < T_SEQ; ++t) {
    const __bf16* hprev = (t & 1) ? hbuf1 : hbuf0;
    __bf16*       hnext = (t & 1) ? hbuf0 : hbuf1;

    // accumulators <- xproj tile (this WG's 4 gate sub-tiles), from LDS
    v8f acc[2][2];
#pragma unroll
    for (int mi = 0; mi < 2; ++mi)
#pragma unroll
      for (int ni = 0; ni < 2; ++ni)
        acc[mi][ni] = load_acc_bf16(xpL + wave * 1024 + (mi * 16) * 32 + ni * 16, 32);
    __syncthreads();                         // everyone done reading xpL

#if HAVE_TDM
    // async prefetch of next step's xproj tile; waited after the grid barrier
    if (wave == 0 && t + 1 < T_SEQ) {
#pragma unroll
      for (int g = 0; g < 4; ++g)
        tdm_load_2d(SC_XPL_OFF + (unsigned)g * 2048u,
                    xproj + (size_t)(t + 1) * BATCH * GATES + (size_t)(g * HIDDEN + hid0),
                    32u, 32u, 32u, 32u, (unsigned)GATES, 1u);
    }
#endif

    // gates[32b x 32u] += h_prev[32 x 512] * WhL[gate-block rows]^T
    for (int k = 0; k < HIDDEN; k += 32) {
      v16bf a0 = load_frag_b(hprev + k, HIDDEN);
      v16bf a1 = load_frag_b(hprev + (size_t)16 * HIDDEN + k, HIDDEN);
      v16bf b0 = load_frag_b(&WhL[(size_t)(wave * 32 + 0) * HIDDEN + k], HIDDEN);
      v16bf b1 = load_frag_b(&WhL[(size_t)(wave * 32 + 16) * HIDDEN + k], HIDDEN);
      acc[0][0] = wmma_bf16(a0, b0, acc[0][0]);
      acc[0][1] = wmma_bf16(a0, b1, acc[0][1]);
      acc[1][0] = wmma_bf16(a1, b0, acc[1][0]);
      acc[1][1] = wmma_bf16(a1, b1, acc[1][1]);
    }

    // scatter gate tiles: gatesL[b][wave*32 + u]
    const int mh = lane >> 4, nn = lane & 15;
#pragma unroll
    for (int mi = 0; mi < 2; ++mi)
#pragma unroll
      for (int ni = 0; ni < 2; ++ni)
#pragma unroll
        for (int r = 0; r < 8; ++r) {
          int b = mi * 16 + mh * 8 + r;
          int u = ni * 16 + nn;
          gatesL[b * 128 + wave * 32 + u] = acc[mi][ni][r];
        }
    __syncthreads();

    // elementwise LSTM cell (8 (b, uj) elements per thread)
#pragma unroll
    for (int i = 0; i < 8; ++i) {
      int b = bquad * 8 + i;
      float fg = sigmoid_f(gatesL[b * 128 +  0 + uj]);
      float ig = sigmoid_f(gatesL[b * 128 + 32 + uj]);
      float gg = tanh_f  (gatesL[b * 128 + 64 + uj]);
      float og = sigmoid_f(gatesL[b * 128 + 96 + uj]);
      float cn = fg * c[i] + ig * gg;
      c[i] = cn;
      float hn = og * tanh_f(cn);
      hnext[b * HIDDEN + hid0 + uj] = (__bf16)hn;
      hsb[(size_t)t * BATCH * HIDDEN + (size_t)b * HIDDEN + hid0 + uj] = (__bf16)hn;
      if (t == T_SEQ - 1) {
        hout[b * HIDDEN + hid0 + uj] = hn;
        cout[b * HIDDEN + hid0 + uj] = cn;
      }
    }

    // grid-wide barrier (monotonic arrive counter)
    __threadfence();
    __syncthreads();
    if (tid == 0) {
      __hip_atomic_fetch_add(arrive, 1u, __ATOMIC_RELEASE, __HIP_MEMORY_SCOPE_AGENT);
      const unsigned target = (unsigned)(NWG * (t + 1));
      while (__hip_atomic_load(arrive, __ATOMIC_ACQUIRE, __HIP_MEMORY_SCOPE_AGENT) < target)
        __builtin_amdgcn_s_sleep(1);
    }
    __syncthreads();

#if HAVE_TDM
    if (wave == 0) __builtin_amdgcn_s_wait_tensorcnt(0);   // next xp tile landed
    __syncthreads();
#else
    if (t + 1 < T_SEQ) {
      for (int idx = tid; idx < 4 * 32 * 32; idx += 128) {
        int g = idx >> 10, b = (idx >> 5) & 31, u = idx & 31;
        xpL[idx] = xproj[(size_t)(t + 1) * BATCH * GATES + (size_t)b * GATES + g * HIDDEN + hid0 + u];
      }
      __syncthreads();
    }
#endif
  }
}

// ---------------------------------------------------------------------------
extern "C" void kernel_launch(void* const* d_in, const int* in_sizes, int n_in,
                              void* d_out, int out_size, void* d_ws, size_t ws_size,
                              hipStream_t stream) {
  const float* x_seq = (const float*)d_in[0];   // [T][B][I]
  const float* Wx    = (const float*)d_in[1];   // [4H][I]
  const float* bx    = (const float*)d_in[2];   // [4H]
  const float* Wh    = (const float*)d_in[3];   // [4H][H]
  const float* Why_w = (const float*)d_in[4];   // [O][H]
  const float* Why_b = (const float*)d_in[5];   // [O]

  float* out  = (float*)d_out;                  // outputs | h | c
  float* hout = out + (size_t)T_SEQ * BATCH * OUTPUT;
  float* cout = hout + (size_t)BATCH * HIDDEN;

  char* ws = (char*)d_ws;                       // needs ~206.2 MB
  __bf16*   xproj  = (__bf16*)(ws + WS_XPROJ);
  __bf16*   xbf    = (__bf16*)(ws + WS_XBF);
  __bf16*   hsb    = (__bf16*)(ws + WS_HSB);
  __bf16*   wxbf   = (__bf16*)(ws + WS_WXBF);
  __bf16*   whbf   = (__bf16*)(ws + WS_WHBF);
  __bf16*   whybf  = (__bf16*)(ws + WS_WHYBF);
  __bf16*   hbuf0  = (__bf16*)(ws + WS_HBUF0);
  __bf16*   hbuf1  = (__bf16*)(ws + WS_HBUF1);
  unsigned* arrive = (unsigned*)(ws + WS_CTRL);

  // 0) bf16 conversions + zero h buffers + barrier counter
  lstm_prep_kernel<<<2048, 256, 0, stream>>>(x_seq, Wx, Wh, Why_w,
                                             xbf, wxbf, whbf, whybf,
                                             hbuf0, hbuf1, arrive);

  // 1) xproj = x @ Wx^T + bx  (M=32768, N=2048, K=512) -> bf16
  gemm_bias_bf16<<<dim3(GATES / 64, (T_SEQ * BATCH) / 128), 128, 8192, stream>>>(
      xbf, wxbf, bx, INPUT, GATES, nullptr, xproj);

  // 2) sequential scan (persistent, 16 WGs, Wh resident in LDS via TDM)
  lstm_scan_kernel<<<NWG, 128, SC_SMEM_SZ, stream>>>(whbf, xproj, hsb,
                                                     hbuf0, hbuf1, arrive, hout, cout);

  // 3) outputs = hs @ Why^T + Why_b  (M=32768, N=512, K=512) -> f32
  gemm_bias_bf16<<<dim3(OUTPUT / 64, (T_SEQ * BATCH) / 128), 128, 8192, stream>>>(
      hsb, whybf, Why_b, HIDDEN, OUTPUT, out, nullptr);
}
